// SutraErrorScratchpad_34084860461136
// MI455X (gfx1250) — compile-verified, matrix operand order checked by reading, not run on hardware
//
#include <hip/hip_runtime.h>
#include <math.h>

// ---------------------------------------------------------------------------
// Sutra error-scratchpad forward for MI455X (gfx1250, wave32, WMMA).
// All large GEMMs: v_wmma_f32_16x16x32_bf16, BK=64 (8 back-to-back WMMAs per
// LDS stage), all B operands pre-transposed to [N][K] bf16 so A and B tiles
// load as contiguous b128 via GLOBAL_LOAD_ASYNC_TO_LDS (ASYNCcnt-tracked).
// ---------------------------------------------------------------------------

#define DIMD 512
#define FFD  2048
#define NSs  6
#define NTOK 2048      // B*T
#define TLEN 1024
#define NSC  8
#define VOC  32000
#define RSQRTD 0.04419417382415922f   // 1/sqrt(512)

typedef __bf16 bf16_t;
typedef bf16_t v16bf __attribute__((ext_vector_type(16)));
typedef bf16_t v8bf  __attribute__((ext_vector_type(8)));
typedef float  v8f   __attribute__((ext_vector_type(8)));
typedef int    v4i_t __attribute__((ext_vector_type(4)));

#if __has_builtin(__builtin_amdgcn_global_load_async_to_lds_b128)
#define HAVE_ASYNC 1
#else
#define HAVE_ASYNC 0
#endif

__device__ __forceinline__ float sigm_(float x){ return 1.f/(1.f+__expf(-x)); }
__device__ __forceinline__ float silu_(float x){ return x*sigm_(x); }
__device__ __forceinline__ float splus_(float x){ return (x>20.f)?x:log1pf(__expf(x)); }
__device__ __forceinline__ bf16_t f2bf(float x){ return (bf16_t)x; }

__device__ __forceinline__ float wave_sum(float v){
#pragma unroll
  for(int off=16; off>0; off>>=1) v += __shfl_xor(v, off, 32);
  return v;
}

// 16-byte global->LDS copy: async (ASYNCcnt) when available, sync fallback.
// Builtin signature (from hipcc): (int4 AS1*, int4 AS3*, imm offset, imm cpol)
__device__ __forceinline__ void cp16(const bf16_t* g, bf16_t* l){
#if HAVE_ASYNC
  __builtin_amdgcn_global_load_async_to_lds_b128(
      (__attribute__((address_space(1))) v4i_t*)g,
      (__attribute__((address_space(3))) v4i_t*)l, 0, 0);
#else
  *(v8bf*)l = *(const v8bf*)g;
#endif
}
__device__ __forceinline__ void cp_wait(){
#if HAVE_ASYNC
#if __has_builtin(__builtin_amdgcn_s_wait_asynccnt)
  __builtin_amdgcn_s_wait_asynccnt(0);
#else
  asm volatile("s_wait_asynccnt 0" ::: "memory");
#endif
#endif
}

// ------------------------------ WMMA GEMM ----------------------------------
// C[M,N] = epilogue(A[M,K](bf16,row-major) @ B(bf16, [N][K] row-major) + bias)
// Block tile 128x64, BK=64. 8 waves, each 32x32 (2x2 wmma tiles, 8 wmma/stage).
#define GM_NONE     0
#define GM_SILU     1
#define GM_SOFTPLUS 2   // softplus(acc+bias)+alpha
#define GM_PI_SCALE 3   // (acc+bias)*pi[row*6+pis]
#define GM_PI_ACCUM 4   // outF += (acc+bias)*pi[row*6+pis]

__global__ __launch_bounds__(256)
void k_gemm(const bf16_t* __restrict__ A, const bf16_t* __restrict__ Bm,
            const float* __restrict__ bias, const float* __restrict__ pi, int pis,
            float* __restrict__ outF, bf16_t* __restrict__ outB,
            int M, int N, int K, int mode, float alpha)
{
  __shared__ __align__(16) bf16_t As[128*64];   // [m][k]
  __shared__ __align__(16) bf16_t Bs[64*64];    // [n][k]
  const int tid  = threadIdx.x;
  const int lane = tid & 31;
  const int wave = tid >> 5;
  const int wm = (wave>>1)*32, wn = (wave&1)*32;
  const int gm0 = blockIdx.y*128, gn0 = blockIdx.x*64;

  v8f c[2][2];
#pragma unroll
  for(int i=0;i<2;i++)
#pragma unroll
    for(int j=0;j<2;j++)
#pragma unroll
      for(int r=0;r<8;r++) c[i][j][r]=0.f;

  const int arow = tid>>1, aseg = (tid&1)*32;   // 128 rows x 64 k : 32 bf16/thread
  const int brow = tid>>2, bseg = (tid&3)*16;   //  64 rows x 64 k : 16 bf16/thread

  for(int k0=0;k0<K;k0+=64){
    { // A tile
      const bf16_t* src = A + (size_t)(gm0+arow)*K + k0 + aseg;
      bf16_t* dst = &As[arow*64 + aseg];
#pragma unroll
      for(int u=0;u<4;u++) cp16(src+u*8, dst+u*8);
      __builtin_prefetch(src+64, 0, 3);        // speculative: next K tile
    }
    { // B tile ([N][K] layout: contiguous along K)
      const bf16_t* src = Bm + (size_t)(gn0+brow)*K + k0 + bseg;
      bf16_t* dst = &Bs[brow*64 + bseg];
#pragma unroll
      for(int u=0;u<2;u++) cp16(src+u*8, dst+u*8);
      __builtin_prefetch(src+64, 0, 3);
    }
    cp_wait();
    __syncthreads();

    // 16-bit A-matrix 16x32 layout (ISA 7.12.2): lanes0-15 rows m, K={0-7,16-23};
    // lanes16-31 same rows, K={8-15,24-31}. B mirrored (column per lane).
    const int kb = (lane<16)?0:8;
    const int mr = lane&15;
    v16bf a[2][2], b[2][2];
#pragma unroll
    for(int h=0;h<2;h++){
      const int kk2 = h*32;
#pragma unroll
      for(int i=0;i<2;i++){
        const bf16_t* p = &As[(wm+i*16+mr)*64 + kk2];
        v8bf lo = *(const v8bf*)(p+kb);
        v8bf hi = *(const v8bf*)(p+16+kb);
#pragma unroll
        for(int e=0;e<8;e++){ a[h][i][e]=lo[e]; a[h][i][e+8]=hi[e]; }
      }
#pragma unroll
      for(int j=0;j<2;j++){
        const bf16_t* p = &Bs[(wn+j*16+mr)*64 + kk2];
        v8bf lo = *(const v8bf*)(p+kb);
        v8bf hi = *(const v8bf*)(p+16+kb);
#pragma unroll
        for(int e=0;e<8;e++){ b[h][j][e]=lo[e]; b[h][j][e+8]=hi[e]; }
      }
    }
#pragma unroll
    for(int h=0;h<2;h++)
#pragma unroll
      for(int i=0;i<2;i++)
#pragma unroll
        for(int j=0;j<2;j++)
          c[i][j] = __builtin_amdgcn_wmma_f32_16x16x32_bf16(
              false, a[h][i], false, b[h][j], (short)0, c[i][j], false, false);
    __syncthreads();
  }

  // C/D 16x16 f32 layout: VGPR r -> row r (lanes0-15) / r+8 (lanes16-31), col=lane%16
  const int col   = lane & 15;
  const int rbase = (lane<16)?0:8;
#pragma unroll
  for(int i=0;i<2;i++){
#pragma unroll
    for(int j=0;j<2;j++){
      int gcol = gn0 + wn + j*16 + col;
      float bv = bias ? bias[gcol] : 0.f;
#pragma unroll
      for(int r=0;r<8;r++){
        int grow = gm0 + wm + i*16 + rbase + r;
        size_t idx = (size_t)grow*N + gcol;
        float acc = c[i][j][r], v;
        if(mode==GM_SILU){ v = silu_(acc+bv); }
        else if(mode==GM_SOFTPLUS){ v = splus_(acc+bv)+alpha; }
        else if(mode==GM_PI_SCALE){ v = (acc+bv)*pi[grow*NSs+pis]; }
        else if(mode==GM_PI_ACCUM){ outF[idx] += (acc+bv)*pi[grow*NSs+pis]; continue; }
        else { v = acc*alpha + bv; }
        if(outF) outF[idx]=v;
        if(outB) outB[idx]=f2bf(v);
      }
    }
  }
}

// ------------------------------ utility kernels ----------------------------
__global__ void k_cvt(const float* __restrict__ s, bf16_t* __restrict__ d, size_t n){
  for(size_t i=(size_t)blockIdx.x*blockDim.x+threadIdx.x; i<n; i+=(size_t)gridDim.x*blockDim.x)
    d[i]=f2bf(s[i]);
}
// fp32 [K][N] -> bf16 [N][K] (weights pre-transposed once per call)
__global__ void k_cvt_t(const float* __restrict__ s, bf16_t* __restrict__ d, int K, int N){
  for(size_t i=(size_t)blockIdx.x*blockDim.x+threadIdx.x; i<(size_t)K*N; i+=(size_t)gridDim.x*blockDim.x){
    int k = (int)(i/(size_t)N), n = (int)(i%(size_t)N);
    d[(size_t)n*K+k]=f2bf(s[i]);
  }
}
__global__ void k_zero(float* p, size_t n){
  for(size_t i=(size_t)blockIdx.x*blockDim.x+threadIdx.x; i<n; i+=(size_t)gridDim.x*blockDim.x)
    p[i]=0.f;
}
__global__ void k_copy(float* d, const float* s, size_t n){
  for(size_t i=(size_t)blockIdx.x*blockDim.x+threadIdx.x; i<n; i+=(size_t)gridDim.x*blockDim.x)
    d[i]=s[i];
}
__global__ void k_embed(const int* __restrict__ x, const float* __restrict__ emb,
                        const float* __restrict__ pos, bf16_t* __restrict__ hB){
  int i = blockIdx.x*blockDim.x + threadIdx.x;
  if(i >= NTOK*DIMD) return;
  int n = i>>9, d = i&511, t = n&(TLEN-1);
  hB[i] = f2bf(emb[(size_t)x[n]*DIMD + d] + pos[(size_t)t*DIMD + d]);
}
__global__ void k_init_pi(float* pi){
  int n = blockIdx.x*blockDim.x+threadIdx.x; if(n>=NTOK) return;
#pragma unroll
  for(int j=0;j<NSs;j++) pi[n*NSs+j] = (j==2)?1.f:0.f;
}
__global__ void k_init_mem(float* mem, const float* mem_init){
  int i = blockIdx.x*blockDim.x+threadIdx.x; if(i>=2*NSC*DIMD) return;
  mem[i]=mem_init[i & (NSC*DIMD-1)];
}
__global__ void k_mu_mean(const float* __restrict__ mu, float* mumean){
  int b=blockIdx.x, d=threadIdx.x;
  float s=0;
  for(int t=0;t<TLEN;t++) s += mu[((size_t)(b*TLEN+t))*DIMD+d];
  mumean[b*DIMD+d]=s*(1.f/TLEN);
}

// transition gating: g=silu(mean@Wg1+bg1); mix=softmax(g@Wg2+bg2); mode=mix@mode_bias
__global__ void k_gate_mode(const float* mumean, const float* Wg1, const float* bg1,
                            const float* Wg2, const float* bg2, const float* mode_bias,
                            float* modeb){
  __shared__ float g[2][32];
  __shared__ float z[2][2];
  int tid = threadIdx.x;
  if(tid<64){
    int b=tid>>5, j=tid&31;
    float a=bg1[j];
    for(int d=0;d<DIMD;d++) a += mumean[b*DIMD+d]*Wg1[d*32+j];
    g[b][j]=silu_(a);
  }
  __syncthreads();
  if(tid<4){
    int b=tid>>1, m=tid&1;
    float a=bg2[m];
    for(int j=0;j<32;j++) a += g[b][j]*Wg2[j*2+m];
    z[b][m]=a;
  }
  __syncthreads();
  if(tid<72){
    int b=tid/36, kk=tid%36;
    float mx=fmaxf(z[b][0],z[b][1]);
    float e0=__expf(z[b][0]-mx), e1=__expf(z[b][1]-mx);
    float inv=1.f/(e0+e1);
    modeb[b*36+kk] = (e0*inv)*mode_bias[kk] + (e1*inv)*mode_bias[36+kk];
  }
}

// base = h1 @ tr_W2 + tr_b2   (N=36, tiny: naive fp32)
__global__ void k_base(const bf16_t* __restrict__ h1b, const float* __restrict__ W2,
                       const float* __restrict__ b2, float* __restrict__ base){
  int i = blockIdx.x*blockDim.x+threadIdx.x; if(i>=NTOK*36) return;
  int n=i/36, j=i%36;
  float a=b2[j];
  for(int k=0;k<2*DIMD;k++) a += (float)h1b[(size_t)n*(2*DIMD)+k]*W2[(size_t)k*36+j];
  base[i]=a;
}

// K = softmax(base+mode) row-wise (6x6); pi_ev = pi @ K
__global__ void k_trans_pi(const float* __restrict__ base, const float* __restrict__ modeb,
                           const float* __restrict__ pi, float* __restrict__ pi_ev){
  int n = blockIdx.x*blockDim.x+threadIdx.x; if(n>=NTOK) return;
  int b = n>>10;
  float Km[NSs][NSs];
  for(int s=0;s<NSs;s++){
    float row[NSs]; float mx=-1e30f;
    for(int u=0;u<NSs;u++){ row[u]=base[(size_t)n*36+s*NSs+u]+modeb[b*36+s*NSs+u]; mx=fmaxf(mx,row[u]); }
    float se=0;
    for(int u=0;u<NSs;u++){ row[u]=__expf(row[u]-mx); se+=row[u]; }
    float inv=1.f/se;
    for(int u=0;u<NSs;u++) Km[s][u]=row[u]*inv;
  }
  for(int u=0;u<NSs;u++){
    float acc=0;
    for(int s=0;s<NSs;s++) acc += pi[(size_t)n*NSs+s]*Km[s][u];
    pi_ev[(size_t)n*NSs+u]=acc;
  }
}

// ev = mu@Wev+bev; pi_new = pi_ev*softmax(2*ev); normalize; top2 project
__global__ void k_ev_pi(const float* __restrict__ mu, const float* __restrict__ Wev,
                        const float* __restrict__ bev, const float* __restrict__ pi_ev,
                        float* __restrict__ pi){
  int n = blockIdx.x*blockDim.x + threadIdx.x; if(n>=NTOK) return;
  float ev[NSs];
#pragma unroll
  for(int j=0;j<NSs;j++) ev[j]=bev[j];
  const float* mp = mu + (size_t)n*DIMD;
  for(int d=0;d<DIMD;d++){
    float m = mp[d];
#pragma unroll
    for(int j=0;j<NSs;j++) ev[j] += m*Wev[d*NSs+j];
  }
  float mx=-1e30f;
#pragma unroll
  for(int j=0;j<NSs;j++) mx=fmaxf(mx, ev[j]*2.f);
  float sm[NSs], se=0;
#pragma unroll
  for(int j=0;j<NSs;j++){ sm[j]=__expf(ev[j]*2.f-mx); se+=sm[j]; }
  float inv=1.f/se;
  float pn[NSs], ps=0;
#pragma unroll
  for(int j=0;j<NSs;j++){ pn[j]=pi_ev[(size_t)n*NSs+j]*sm[j]*inv; ps+=pn[j]; }
  float q=1.f/fmaxf(ps,1e-8f);
#pragma unroll
  for(int j=0;j<NSs;j++) pn[j]*=q;
  int i1=0;
#pragma unroll
  for(int j=1;j<NSs;j++) if(pn[j]>pn[i1]) i1=j;
  int i2=(i1==0)?1:0;
#pragma unroll
  for(int j=0;j<NSs;j++) if(j!=i1 && pn[j]>pn[i2]) i2=j;
  float q2 = 1.f/fmaxf(pn[i1]+pn[i2],1e-8f);
#pragma unroll
  for(int j=0;j<NSs;j++) pi[(size_t)n*NSs+j] = (j==i1||j==i2)? pn[j]*q2 : 0.f;
}

// windowed attention router: 16 positions, top-4, softmax, weighted V sum.
__global__ void k_router(const float* __restrict__ q, const float* __restrict__ k,
                         const float* __restrict__ v, float* __restrict__ msgr){
  int wave = threadIdx.x>>5, lane = threadIdx.x&31;
  int n = blockIdx.x*8 + wave; if(n>=NTOK) return;
  int b = n>>10, t = n&1023;
  const float* qp = q + (size_t)n*DIMD;
  float sc[16];
#pragma unroll
  for(int w=0;w<16;w++){
    int src = t-15+w;
    float s = -1e9f;
    if(src>=0){
      const float* kp = k + (size_t)(b*TLEN+src)*DIMD;
      float acc=0;
      for(int d=lane; d<DIMD; d+=32) acc += qp[d]*kp[d];
      s = wave_sum(acc)*RSQRTD;
    }
    sc[w]=s;
  }
  int idx4[4]; float val[4]; unsigned used=0;
#pragma unroll
  for(int kk=0;kk<4;kk++){
    float best=-1e30f; int bi=0;
    for(int w=0;w<16;w++)
      if(!((used>>w)&1u) && sc[w]>best){ best=sc[w]; bi=w; }
    used |= 1u<<bi; idx4[kk]=bi; val[kk]=best;
  }
  float mx=val[0], se=0, at[4];
#pragma unroll
  for(int kk=0;kk<4;kk++){ at[kk]=__expf(val[kk]-mx); se+=at[kk]; }
  float inv=1.f/se;
#pragma unroll
  for(int kk=0;kk<4;kk++) at[kk]*=inv;
  for(int d=lane; d<DIMD; d+=32){
    float m=0;
#pragma unroll
    for(int kk=0;kk<4;kk++){
      int src=t-15+idx4[kk];
      if(src>=0) m += at[kk]*v[(size_t)(b*TLEN+src)*DIMD+d];
    }
    msgr[(size_t)n*DIMD+d]=m;
  }
}

// scratch read + combine: msgs += 0.1*pi3*softmax(q2·mem/sqrtD)·mem ; write bf16
__global__ void k_sread(const float* __restrict__ q2, const float* __restrict__ mem,
                        const float* __restrict__ pi, float* __restrict__ msgs,
                        bf16_t* __restrict__ msgsB){
  int wave=threadIdx.x>>5, lane=threadIdx.x&31;
  int n = blockIdx.x*8+wave; if(n>=NTOK) return;
  int b=n>>10;
  const float* qp = q2 + (size_t)n*DIMD;
  float sc[NSC];
#pragma unroll
  for(int s=0;s<NSC;s++){
    const float* mp = mem + (size_t)(b*NSC+s)*DIMD;
    float acc=0;
    for(int d=lane; d<DIMD; d+=32) acc += qp[d]*mp[d];
    sc[s]=wave_sum(acc)*RSQRTD;
  }
  float mx=-1e30f;
#pragma unroll
  for(int s=0;s<NSC;s++) mx=fmaxf(mx,sc[s]);
  float se=0;
#pragma unroll
  for(int s=0;s<NSC;s++){ sc[s]=__expf(sc[s]-mx); se+=sc[s]; }
  float inv=1.f/se;
#pragma unroll
  for(int s=0;s<NSC;s++) sc[s]*=inv;
  float p3 = pi[(size_t)n*NSs+3];
  for(int d=lane; d<DIMD; d+=32){
    float ctx=0;
#pragma unroll
    for(int s=0;s<NSC;s++) ctx += sc[s]*mem[(size_t)(b*NSC+s)*DIMD+d];
    size_t i=(size_t)n*DIMD+d;
    float val = msgs[i] + 0.1f*p3*ctx;
    msgs[i]=val; msgsB[i]=f2bf(val);
  }
}

// Bayesian write + stage-bank residual: mu=(lam*mu+w*obs*msg)/lam' + 0.1*so
__global__ void k_bayes(float* __restrict__ mu, float* __restrict__ lam,
                        bf16_t* __restrict__ muB, const float* __restrict__ msgs,
                        const float* __restrict__ obs, const float* __restrict__ so,
                        const float* __restrict__ pi){
  size_t i = (size_t)blockIdx.x*blockDim.x + threadIdx.x;
  if(i >= (size_t)NTOK*DIMD) return;
  int n = (int)(i>>9);
  float w = pi[(size_t)n*NSs+4];
  float ol = obs[i], l = lam[i], m = mu[i];
  float ln2 = l + w*ol;
  float mn = (l*m + w*ol*msgs[i]) / ln2;
  mn += 0.1f*so[i];
  mu[i]=mn; lam[i]=ln2; muB[i]=f2bf(mn);
}

// delta = LN(mu - mu_prev); accumulate pi_w-weighted sums per batch
__global__ void k_delta_ws(const float* __restrict__ mu, const float* __restrict__ mup,
                           const float* __restrict__ pi, const float* __restrict__ eg,
                           const float* __restrict__ eb, float* wsnum, float* wsden){
  int wave = threadIdx.x>>5, lane = threadIdx.x&31;
  int n = blockIdx.x*8 + wave; if(n>=NTOK) return;
  int b = n>>10;
  const float* a  = mu  + (size_t)n*DIMD;
  const float* p2 = mup + (size_t)n*DIMD;
  float xl[16], s=0;
#pragma unroll
  for(int i=0;i<16;i++){ int d=lane+32*i; float x=a[d]-p2[d]; xl[i]=x; s+=x; }
  float mean = wave_sum(s)*(1.f/DIMD);
  float v=0;
#pragma unroll
  for(int i=0;i<16;i++){ float t=xl[i]-mean; v+=t*t; }
  float inv = rsqrtf(wave_sum(v)*(1.f/DIMD)+1e-5f);
  float pw = pi[(size_t)n*NSs+4];
#pragma unroll
  for(int i=0;i<16;i++){
    int d=lane+32*i;
    float delta=(xl[i]-mean)*inv*eg[d]+eb[d];
    atomicAdd(&wsnum[b*DIMD+d], delta*pw);
  }
  if(lane==0) atomicAdd(&wsden[b], pw);
}

// mem' = EMA*mem + (1-EMA)*sigmoid([mem,ws]@wg)*(ws@wv+bv)   (ping-pong buffers)
__global__ void k_swrite(const float* __restrict__ mem, float* __restrict__ memN,
                         const float* __restrict__ wsnum, const float* __restrict__ wsden,
                         const float* __restrict__ wg_W, const float* __restrict__ wg_b,
                         const float* __restrict__ wv_W, const float* __restrict__ wv_b){
  int i = blockIdx.x*blockDim.x+threadIdx.x;
  if(i>=2*NSC*DIMD) return;
  int d=i&511, slot=(i>>9)&7, b=i>>12;
  float dinv = 1.f/fmaxf(wsden[b],1e-6f);
  const float* mr = mem + (size_t)(b*NSC+slot)*DIMD;
  float acc = wg_b[d];
  for(int k=0;k<DIMD;k++) acc += mr[k]*wg_W[(size_t)k*DIMD+d];
  for(int k=0;k<DIMD;k++) acc += (wsnum[b*DIMD+k]*dinv)*wg_W[(size_t)(DIMD+k)*DIMD+d];
  float gate = sigm_(acc);
  float nv = wv_b[d];
  for(int k=0;k<DIMD;k++) nv += (wsnum[b*DIMD+k]*dinv)*wv_W[(size_t)k*DIMD+d];
  memN[i] = 0.9f*mr[d] + 0.1f*gate*nv;
}

__global__ void k_final_ln(const float* __restrict__ mu, const float* __restrict__ g,
                           const float* __restrict__ bb, bf16_t* __restrict__ outB){
  int wave = threadIdx.x>>5, lane = threadIdx.x&31;
  int n = blockIdx.x*8 + wave; if(n>=NTOK) return;
  const float* a = mu + (size_t)n*DIMD;
  float xl[16], s=0;
#pragma unroll
  for(int i=0;i<16;i++){ int d=lane+32*i; xl[i]=a[d]; s+=xl[i]; }
  float mean = wave_sum(s)*(1.f/DIMD);
  float v=0;
#pragma unroll
  for(int i=0;i<16;i++){ float t=xl[i]-mean; v+=t*t; }
  float inv = rsqrtf(wave_sum(v)*(1.f/DIMD)+1e-5f);
#pragma unroll
  for(int i=0;i<16;i++){
    int d=lane+32*i;
    outB[(size_t)n*DIMD+d]=f2bf((xl[i]-mean)*inv*g[d]+bb[d]);
  }
}

// ---------------------------------------------------------------------------
extern "C" void kernel_launch(void* const* d_in, const int* in_sizes, int n_in,
                              void* d_out, int out_size, void* d_ws, size_t ws_size,
                              hipStream_t stream){
  (void)in_sizes; (void)n_in; (void)out_size; (void)ws_size;
  const int*   x        = (const int*)  d_in[0];
  const float* emb      = (const float*)d_in[1];
  const float* pos      = (const float*)d_in[2];
  const float* imu_b    = (const float*)d_in[4];
  const float* ilam_b   = (const float*)d_in[6];
  const float* tr_b1    = (const float*)d_in[8];
  const float* tr_W2    = (const float*)d_in[9];
  const float* tr_b2    = (const float*)d_in[10];
  const float* tr_Wg1   = (const float*)d_in[11];
  const float* tr_bg1   = (const float*)d_in[12];
  const float* tr_Wg2   = (const float*)d_in[13];
  const float* tr_bg2   = (const float*)d_in[14];
  const float* mode_bias= (const float*)d_in[15];
  const float* sb_b1    = (const float*)d_in[17];
  const float* sb_b2    = (const float*)d_in[19];
  const float* sb_Wev   = (const float*)d_in[20];
  const float* sb_bev   = (const float*)d_in[21];
  const float* rt_bq    = (const float*)d_in[23];
  const float* rt_bk    = (const float*)d_in[25];
  const float* rt_bv    = (const float*)d_in[27];
  const float* rt_bo    = (const float*)d_in[29];
  const float* wr_b     = (const float*)d_in[31];
  const float* mem_init = (const float*)d_in[32];
  const float* rd_b     = (const float*)d_in[34];
  const float* wg_W     = (const float*)d_in[35];
  const float* wg_b     = (const float*)d_in[36];
  const float* wv_W     = (const float*)d_in[37];
  const float* wv_b     = (const float*)d_in[38];
  const float* eln_g    = (const float*)d_in[39];
  const float* eln_b    = (const float*)d_in[40];
  const float* ln_g     = (const float*)d_in[41];
  const float* ln_b     = (const float*)d_in[42];

  size_t off = 0;
  auto alloc = [&](size_t bytes)->void*{
    off = (off + 255) & ~(size_t)255;
    void* p = (char*)d_ws + off; off += bytes; return p; };

  // bf16 weight mirrors, all stored [N][K] (fit in L2: ~64MB)
  bf16_t* embB  = (bf16_t*)alloc((size_t)VOC*DIMD*2);       // emb already [V][D]
  bf16_t* imuB  = (bf16_t*)alloc((size_t)DIMD*DIMD*2);
  bf16_t* ilamB = (bf16_t*)alloc((size_t)DIMD*DIMD*2);
  bf16_t* trW1B = (bf16_t*)alloc((size_t)DIMD*2*DIMD*2);
  bf16_t* sbW1B = (bf16_t*)alloc((size_t)NSs*DIMD*FFD*2);
  bf16_t* sbW2B = (bf16_t*)alloc((size_t)NSs*FFD*DIMD*2);
  bf16_t* rtWqB = (bf16_t*)alloc((size_t)DIMD*DIMD*2);
  bf16_t* rtWkB = (bf16_t*)alloc((size_t)DIMD*DIMD*2);
  bf16_t* rtWvB = (bf16_t*)alloc((size_t)DIMD*DIMD*2);
  bf16_t* rtWoB = (bf16_t*)alloc((size_t)DIMD*DIMD*2);
  bf16_t* wrWB  = (bf16_t*)alloc((size_t)DIMD*DIMD*2);
  bf16_t* rdWB  = (bf16_t*)alloc((size_t)DIMD*DIMD*2);
  // activations
  bf16_t* hB     = (bf16_t*)alloc((size_t)NTOK*DIMD*2);
  float*  muF    = (float*) alloc((size_t)NTOK*DIMD*4);
  bf16_t* muB_   = (bf16_t*)alloc((size_t)NTOK*DIMD*2);
  float*  lamF   = (float*) alloc((size_t)NTOK*DIMD*4);
  float*  muPrev = (float*) alloc((size_t)NTOK*DIMD*4);
  float*  piF    = (float*) alloc((size_t)NTOK*NSs*4);
  float*  piEv   = (float*) alloc((size_t)NTOK*NSs*4);
  float*  mem0   = (float*) alloc((size_t)2*NSC*DIMD*4);
  float*  mem1   = (float*) alloc((size_t)2*NSC*DIMD*4);
  float*  mumean = (float*) alloc((size_t)2*DIMD*4);
  float*  modeb  = (float*) alloc((size_t)2*36*4);
  bf16_t* h1B    = (bf16_t*)alloc((size_t)NTOK*2*DIMD*2);
  float*  baseF  = (float*) alloc((size_t)NTOK*36*4);
  bf16_t* hsB    = (bf16_t*)alloc((size_t)NTOK*FFD*2);
  float*  soF    = (float*) alloc((size_t)NTOK*DIMD*4);
  float*  qF     = (float*) alloc((size_t)NTOK*DIMD*4);
  float*  kF     = (float*) alloc((size_t)NTOK*DIMD*4);
  float*  vF     = (float*) alloc((size_t)NTOK*DIMD*4);
  float*  msgrF  = (float*) alloc((size_t)NTOK*DIMD*4);
  bf16_t* msgrB  = (bf16_t*)alloc((size_t)NTOK*DIMD*2);
  float*  msgsF  = (float*) alloc((size_t)NTOK*DIMD*4);
  bf16_t* msgsB  = (bf16_t*)alloc((size_t)NTOK*DIMD*2);
  float*  q2F    = (float*) alloc((size_t)NTOK*DIMD*4);
  float*  obsF   = (float*) alloc((size_t)NTOK*DIMD*4);
  float*  wsnum  = (float*) alloc((size_t)(2*DIMD+2)*4);
  float*  wsden  = wsnum + 2*DIMD;
  bf16_t* finalB = (bf16_t*)alloc((size_t)NTOK*DIMD*2);

  auto cvtT = [&](const float* s, bf16_t* d, int K, int N){
    k_cvt_t<<<2048,256,0,stream>>>(s,d,K,N); };
  auto gemm = [&](const bf16_t* A, const bf16_t* Bm, const float* bias,
                  const float* pip, int pis, float* oF, bf16_t* oB,
                  int M, int N, int K, int mode, float alpha){
    dim3 g(N/64, M/128);
    k_gemm<<<g, 256, 0, stream>>>(A,Bm,bias,pip,pis,oF,oB,M,N,K,mode,alpha); };

  // weight casts (+transpose to [N][K]) — deterministic per call
  k_cvt<<<2048,256,0,stream>>>(emb, embB, (size_t)VOC*DIMD);
  cvtT((const float*)d_in[3],  imuB,  DIMD, DIMD);
  cvtT((const float*)d_in[5],  ilamB, DIMD, DIMD);
  cvtT((const float*)d_in[7],  trW1B, DIMD, 2*DIMD);
  for(int s=0;s<NSs;s++){
    cvtT((const float*)d_in[16] + (size_t)s*DIMD*FFD, sbW1B + (size_t)s*DIMD*FFD, DIMD, FFD);
    cvtT((const float*)d_in[18] + (size_t)s*FFD*DIMD, sbW2B + (size_t)s*FFD*DIMD, FFD, DIMD);
  }
  cvtT((const float*)d_in[22], rtWqB, DIMD, DIMD);
  cvtT((const float*)d_in[24], rtWkB, DIMD, DIMD);
  cvtT((const float*)d_in[26], rtWvB, DIMD, DIMD);
  cvtT((const float*)d_in[28], rtWoB, DIMD, DIMD);
  cvtT((const float*)d_in[30], wrWB,  DIMD, DIMD);
  cvtT((const float*)d_in[33], rdWB,  DIMD, DIMD);

  // init
  k_embed<<<4096,256,0,stream>>>(x, emb, pos, hB);
  gemm(hB, imuB,  imu_b,  nullptr,0, muF,  muB_,   NTOK, DIMD, DIMD, GM_NONE, 1.f);
  gemm(hB, ilamB, ilam_b, nullptr,0, lamF, nullptr,NTOK, DIMD, DIMD, GM_SOFTPLUS, 0.1f);
  k_copy<<<2048,256,0,stream>>>(muPrev, muF, (size_t)NTOK*DIMD);
  k_init_pi<<<8,256,0,stream>>>(piF);
  k_init_mem<<<32,256,0,stream>>>(mem0, mem_init);
  float *mcur=mem0, *mnext=mem1;

  for(int step=0; step<6; step++){
    // transition
    k_mu_mean<<<2,512,0,stream>>>(muF, mumean);
    k_gate_mode<<<1,128,0,stream>>>(mumean, tr_Wg1, tr_bg1, tr_Wg2, tr_bg2, mode_bias, modeb);
    gemm(muB_, trW1B, tr_b1, nullptr,0, nullptr, h1B, NTOK, 2*DIMD, DIMD, GM_SILU, 0.f);
    k_base<<<288,256,0,stream>>>(h1B, tr_W2, tr_b2, baseF);
    k_trans_pi<<<8,256,0,stream>>>(baseF, modeb, piF, piEv);
    // stage bank (uses old pi)
    k_zero<<<2048,256,0,stream>>>(soF, (size_t)NTOK*DIMD);
    for(int s=0;s<NSs;s++){
      gemm(muB_, sbW1B + (size_t)s*DIMD*FFD, sb_b1 + s*FFD, nullptr,0,
           nullptr, hsB, NTOK, FFD, DIMD, GM_SILU, 0.f);
      gemm(hsB, sbW2B + (size_t)s*FFD*DIMD, sb_b2 + s*DIMD, piF, s,
           soF, nullptr, NTOK, DIMD, FFD, GM_PI_ACCUM, 0.f);
    }
    // pi update (ev + top2)
    k_ev_pi<<<8,256,0,stream>>>(muF, sb_Wev, sb_bev, piEv, piF);
    // router
    gemm(muB_, rtWqB, rt_bq, nullptr,0, qF, nullptr, NTOK, DIMD, DIMD, GM_NONE, 1.f);
    gemm(muB_, rtWkB, rt_bk, nullptr,0, kF, nullptr, NTOK, DIMD, DIMD, GM_NONE, 1.f);
    gemm(muB_, rtWvB, rt_bv, nullptr,0, vF, nullptr, NTOK, DIMD, DIMD, GM_NONE, 1.f);
    k_router<<<256,256,0,stream>>>(qF,kF,vF,msgrF);
    k_cvt<<<2048,256,0,stream>>>(msgrF, msgrB, (size_t)NTOK*DIMD);
    gemm(msgrB, rtWoB, rt_bo, piF, 3, msgsF, nullptr, NTOK, DIMD, DIMD, GM_PI_SCALE, 0.f);
    // scratch read
    gemm(muB_, rdWB, rd_b, nullptr,0, q2F, nullptr, NTOK, DIMD, DIMD, GM_NONE, 1.f);
    k_sread<<<256,256,0,stream>>>(q2F, mcur, piF, msgsF, msgsB);
    // bayes write
    gemm(msgsB, wrWB, wr_b, nullptr,0, obsF, nullptr, NTOK, DIMD, DIMD, GM_SOFTPLUS, 1e-4f);
    k_bayes<<<4096,256,0,stream>>>(muF, lamF, muB_, msgsF, obsF, soF, piF);
    // scratch write (ping-pong mem)
    k_zero<<<8,256,0,stream>>>(wsnum, (size_t)(2*DIMD+2));
    k_delta_ws<<<256,256,0,stream>>>(muF, muPrev, piF, eln_g, eln_b, wsnum, wsden);
    k_swrite<<<32,256,0,stream>>>(mcur, mnext, wsnum, wsden, wg_W, wg_b, wv_W, wv_b);
    k_copy<<<2048,256,0,stream>>>(muPrev, muF, (size_t)NTOK*DIMD);
    float* t=mcur; mcur=mnext; mnext=t;
  }

  // logits = LN(mu) @ emb^T / sqrt(D)   (emb is [V][D] = [N][K] already)
  k_final_ln<<<256,256,0,stream>>>(muF, ln_g, ln_b, finalB);
  gemm(finalB, embB, nullptr, nullptr,0, (float*)d_out, nullptr,
       NTOK, VOC, DIMD, GM_NONE, RSQRTD);
}
